// ManhattanAttention_44263932952676
// MI455X (gfx1250) — matrix operand order repaired and checked by reading, not compiled, not run on hardware
//
#include <hip/hip_runtime.h>
#include <hip/hip_bf16.h>

// Problem constants from the reference (fixed).
#define BB 2
#define TT 1024
#define DD 128
#define NEG_INF (-1.0e9f)

typedef __attribute__((ext_vector_type(16))) __bf16 v16bf;
typedef __attribute__((ext_vector_type(8)))  float  v8f;
typedef int v4i __attribute__((vector_size(16)));   // int32x4 for b128 async copies

#if defined(__has_builtin)
#if __has_builtin(__builtin_amdgcn_global_load_async_to_lds_b128) && \
    __has_builtin(__builtin_amdgcn_s_wait_asynccnt)
#define HAVE_ASYNC_LDS 1
#endif
#endif

#ifdef HAVE_ASYNC_LDS
// Builtin signature (from hipcc diagnostic): (v4i __device__*, v4i __shared__*, int, int)
#define GAS_PTR(p) ((__attribute__((address_space(1))) v4i*)(p))
#define LAS_PTR(p) ((__attribute__((address_space(3))) v4i*)(p))
#endif

__device__ __forceinline__ float wave_max32(float v) {
#pragma unroll
  for (int o = 16; o > 0; o >>= 1) v = fmaxf(v, __shfl_xor(v, o, 32));
  return v;
}
__device__ __forceinline__ float wave_sum32(float v) {
#pragma unroll
  for (int o = 16; o > 0; o >>= 1) v += __shfl_xor(v, o, 32);
  return v;
}

// float -> bf16 round-to-nearest-even
__device__ __forceinline__ __bf16 f2bf(float f) {
  union { float f; unsigned u; } v; v.f = f;
  unsigned r = (v.u + 0x7FFFu + ((v.u >> 16) & 1u)) >> 16;
  union { unsigned short s; __bf16 b; } o; o.s = (unsigned short)r;
  return o.b;
}

// ---------------------------------------------------------------------------
// Kernel 1: wj[b,t] = dot(h[b,t,:], w)  (one wave32 per row, shuffle reduce)
//           also writes out[b,t,0:128] = h[b,t,:]  (concat's first half)
// ---------------------------------------------------------------------------
__global__ void manh_wj_kernel(const float* __restrict__ h,
                               const float* __restrict__ w,
                               float* __restrict__ out,
                               float* __restrict__ wj) {
  const int wave = threadIdx.x >> 5;
  const int lane = threadIdx.x & 31;
  const int row  = blockIdx.x * 8 + wave;          // [0, B*T)
  const float* hrow = h + (size_t)row * DD;
  float vals[4];
  float acc = 0.f;
#pragma unroll
  for (int q = 0; q < 4; ++q) {
    const int d = lane + 32 * q;
    const float hv = hrow[d];
    vals[q] = hv;
    acc += hv * w[d];
  }
  acc = wave_sum32(acc);
  if (lane == 0) wj[row] = acc;
  float* orow = out + (size_t)row * (2 * DD);
#pragma unroll
  for (int q = 0; q < 4; ++q) orow[lane + 32 * q] = vals[q];
}

// ---------------------------------------------------------------------------
// Kernel 2: flash-style masked softmax attention with L1-distance scores.
// Block = 16 i-rows, 256 threads (8 waves). Streams j in chunks of 32 rows
// staged into LDS via gfx1250 async global->LDS b128 copies. ctx accumulated
// with v_wmma_f32_16x16x32_bf16; the 8 waves split the 128 output columns.
// ---------------------------------------------------------------------------
__global__ void __launch_bounds__(256)
manh_attn_kernel(const float* __restrict__ h,
                 const int* __restrict__ mask,
                 const float* __restrict__ wj,
                 float* __restrict__ out) {
  const int b    = blockIdx.x >> 6;          // T/16 = 64 i-tiles per batch
  const int i0   = (blockIdx.x & 63) * 16;
  const int tid  = threadIdx.x;
  const int lane = tid & 31;
  const int wave = tid >> 5;

  // LDS (row-padded +4 floats: pitch 132 floats = 528B, 16B aligned, bank-rotating)
  __shared__ alignas(16) float  s_hi[16][DD + 4];   // i-tile, fp32
  __shared__ alignas(16) float  s_hj[32][DD + 4];   // j-chunk, fp32
  __shared__ float  s_sc[16][32];                   // score tile
  __shared__ __bf16 s_al[16][32];                   // alpha tile (WMMA A source)
  __shared__ float  s_m[16], s_l[16], s_scale[16];

  const float* hb = h + (size_t)b * TT * DD;

  // stage h_i tile (coalesced, synchronous; read many times, done once)
#pragma unroll
  for (int q = 0; q < 8; ++q) {
    const int idx = q * 256 + tid;
    const int r = idx >> 7, d = idx & 127;
    s_hi[r][d] = hb[(size_t)(i0 + r) * DD + d];
  }
  if (tid < 16) { s_m[tid] = -3.0e38f; s_l[tid] = 0.f; }

  // score-phase pair indices: this thread owns (pi, pj) and (pi, pj+16)
  const int pi = tid & 15;
  const int pj = tid >> 4;

  // WMMA fragment indices (CDNA5 16x16 layouts)
  const int half = lane >> 4;     // lane-half selects K/M split
  const int nco  = lane & 15;     // N (column) / M (for A) within tile
  v8f c = {};                     // fp32 accumulator, 16x16 C/D layout

  __syncthreads();

  for (int jc = 0; jc < TT / 32; ++jc) {
    const int jbase = jc * 32;

    // 1. stage h_j chunk: 32x128 fp32 = 16KB. Each thread moves 4 x 16B.
#ifdef HAVE_ASYNC_LDS
#pragma unroll
    for (int q = 0; q < 4; ++q) {
      const int idx = (q * 256 + tid) * 4;          // float index in chunk
      const int r = idx >> 7, d = idx & 127;
      __builtin_amdgcn_global_load_async_to_lds_b128(
          GAS_PTR(&hb[(size_t)(jbase + r) * DD + d]),
          LAS_PTR(&s_hj[r][d]), 0, 0);
    }
    __builtin_amdgcn_s_wait_asynccnt(0);
#else
#pragma unroll
    for (int q = 0; q < 16; ++q) {
      const int idx = q * 256 + tid;
      const int r = idx >> 7, d = idx & 127;
      s_hj[r][d] = hb[(size_t)(jbase + r) * DD + d];
    }
#endif
    if (jc + 1 < TT / 32)
      __builtin_prefetch(&hb[(size_t)(jbase + 32) * DD + tid * 16], 0, 1);
    __syncthreads();

    // 2. 16x32 score tile: L1 distance from LDS (float4 reads), mask + -wj scale
    float d0 = 0.f, d1 = 0.f;
    {
      const float4* hi4 = (const float4*)&s_hi[pi][0];
      const float4* hx4 = (const float4*)&s_hj[pj][0];
      const float4* hy4 = (const float4*)&s_hj[pj + 16][0];
#pragma unroll 8
      for (int q = 0; q < DD / 4; ++q) {
        const float4 a = hi4[q];
        const float4 x = hx4[q];
        const float4 y = hy4[q];
        d0 += fabsf(a.x - x.x) + fabsf(a.y - x.y) +
              fabsf(a.z - x.z) + fabsf(a.w - x.w);
        d1 += fabsf(a.x - y.x) + fabsf(a.y - y.y) +
              fabsf(a.z - y.z) + fabsf(a.w - y.w);
      }
    }
    const int j0 = jbase + pj, j1 = j0 + 16;
    s_sc[pi][pj]      = mask[b * TT + j0] ? (-wj[b * TT + j0] * d0) : NEG_INF;
    s_sc[pi][pj + 16] = mask[b * TT + j1] ? (-wj[b * TT + j1] * d1) : NEG_INF;
    __syncthreads();

    // 3. online softmax: wave w owns rows 2w, 2w+1 (32 scores each)
#pragma unroll
    for (int rr = 0; rr < 2; ++rr) {
      const int i = wave * 2 + rr;
      const float s    = s_sc[i][lane];
      const float cmax = wave_max32(s);
      const float mo   = s_m[i];
      const float mn   = fmaxf(mo, cmax);
      const float pl   = __expf(s - mn);
      const float cs   = wave_sum32(pl);
      const float scl  = __expf(mo - mn);
      if (lane == 0) { s_m[i] = mn; s_l[i] = s_l[i] * scl + cs; s_scale[i] = scl; }
      s_al[i][lane] = f2bf(pl);
    }
    __syncthreads();

    // 4. rescale accumulator, then WMMA: ctx += alpha(16x32) @ hj(32x16 per wave)
#pragma unroll
    for (int r = 0; r < 8; ++r) c[r] *= s_scale[r + 8 * half];

    // A fragment: 16x32 bf16. Lanes 0-15 hold K=0..15, lanes 16-31
    // hold K=16..31; VGPR p packs one K pair.
    v16bf afrag;
#pragma unroll
    for (int p = 0; p < 8; ++p) {
      const int k = half * 16 + 2 * p;
      afrag[2 * p]     = s_al[nco][k];
      afrag[2 * p + 1] = s_al[nco][k + 1];
    }

    // B fragment: 32x16 bf16. Lanes 0-15 hold K=0..15, lanes 16-31 K=16..31;
    // VGPR p packs one K pair for column N = 16*wave + nco.
    v16bf bfrag;
    const int col = wave * 16 + nco;
#pragma unroll
    for (int p = 0; p < 8; ++p) {
      const int k = half * 16 + 2 * p;
      bfrag[2 * p]     = f2bf(s_hj[k][col]);
      bfrag[2 * p + 1] = f2bf(s_hj[k + 1][col]);
    }

    c = __builtin_amdgcn_wmma_f32_16x16x32_bf16(false, afrag, false, bfrag,
                                                (short)0, c, false, false);
    __syncthreads();
  }

  // epilogue: ctx row i = r + 8*half, col = 16*wave + nco; divide by softmax sum
#pragma unroll
  for (int r = 0; r < 8; ++r) {
    const int i = r + 8 * half;
    const float v = c[r] / s_l[i];
    out[(size_t)(b * TT + i0 + i) * (2 * DD) + DD + wave * 16 + nco] = v;
  }
}

// ---------------------------------------------------------------------------
extern "C" void kernel_launch(void* const* d_in, const int* in_sizes, int n_in,
                              void* d_out, int out_size, void* d_ws, size_t ws_size,
                              hipStream_t stream) {
  (void)in_sizes; (void)n_in; (void)out_size; (void)ws_size;
  const float* h    = (const float*)d_in[0];   // [B,T,D] fp32
  const int*   mask = (const int*)d_in[1];     // [B,T]   bool->int
  const float* w    = (const float*)d_in[2];   // [D]     fp32
  float*       out  = (float*)d_out;           // [B,T,2D] fp32
  float*       wj   = (float*)d_ws;            // [B*T]   fp32 scratch

  // Kernel 1: one wave per row -> 2048 waves = 256 blocks x 8 waves
  manh_wj_kernel<<<(BB * TT) / 8, 256, 0, stream>>>(h, w, out, wj);

  // Kernel 2: one block per (b, 16-row i-tile) -> 128 blocks x 256 threads
  manh_attn_kernel<<<BB * (TT / 16), 256, 0, stream>>>(h, mask, wj, out);
}
